// RCAModel_13065290515064
// MI455X (gfx1250) — matrix operand order repaired and checked by reading, not compiled, not run on hardware
//
#include <hip/hip_runtime.h>
#include <hip/hip_bf16.h>
#include <stdint.h>

typedef __attribute__((ext_vector_type(16))) __bf16 v16bf;
typedef __attribute__((ext_vector_type(8)))  float  v8f;

#define B_   256
#define S_   256
#define D_   2048
#define H1_  1024
#define H2_  512

static __device__ __forceinline__ uint32_t f2bf(float f) {
    // round-to-nearest-even f32 -> bf16 (returned in low 16 bits)
    uint32_t u = __float_as_uint(f);
    u += 0x7FFFu + ((u >> 16) & 1u);
    return u >> 16;
}

// ---------------------------------------------------------------------------
// Prep kernels: factor GEMM1 through the concat structure, pre-swizzle w2.
// ---------------------------------------------------------------------------
__global__ void prep_pre1(const float* __restrict__ fused, const float* __restrict__ w1,
                          const float* __restrict__ b1, float* __restrict__ pre1) {
    int id = blockIdx.x * blockDim.x + threadIdx.x;   // B_*H1_ threads
    int b = id >> 10, h = id & 1023;                  // b uniform per block
    float s = b1[h];
    const float* fr = fused + b * D_;
    for (int k = 0; k < D_; ++k) s = fmaf(fr[k], w1[k * H1_ + h], s);
    pre1[id] = s;
}

__global__ void prep_pre2(const float* __restrict__ emb, const float* __restrict__ w1,
                          float* __restrict__ pre2) {
    int id = blockIdx.x * blockDim.x + threadIdx.x;   // S_*H1_ threads
    int s = id >> 10, h = id & 1023;
    float acc = 0.f;
    const float* er = emb + s * 64;
    const float* wb = w1 + (size_t)D_ * H1_;          // rows 2048..2111
    for (int k = 0; k < 64; ++k) acc = fmaf(er[k], wb[k * H1_ + h], acc);
    pre2[id] = acc;
}

// w2 (1024x512 f32, row-major) -> bf16 packed in this file's B-fragment order (contiguous k, same as A):
// word index = ((kc*32 + nt)*32 + lane)*8 + v
//   n = nt*16 + (lane&15); K = kc*32 + (lane>>4)*16 + 2v; word = {bf(K+1,n),bf(K,n)}
__global__ void prep_w2frag(const float* __restrict__ w2, uint32_t* __restrict__ w2f) {
    int idx = blockIdx.x * blockDim.x + threadIdx.x;  // 262144 threads
    int v = idx & 7, lane = (idx >> 3) & 31, nt = (idx >> 8) & 31, kc = idx >> 13;
    int n = nt * 16 + (lane & 15);
    int K = kc * 32 + ((lane >> 4) << 4) + 2 * v;
    uint32_t lo = f2bf(w2[(size_t)K * H2_ + n]);
    uint32_t hi = f2bf(w2[(size_t)(K + 1) * H2_ + n]);
    w2f[idx] = lo | (hi << 16);
}

// ---------------------------------------------------------------------------
// Main fused kernel: per 16-row tile (fixed b, 16 s values):
//   LN1+relu -> bf16 A-frags in LDS -> WMMA GEMM vs w2 -> LN2+relu -> w3 dot
// ---------------------------------------------------------------------------
__global__ __launch_bounds__(256, 2) void rca_main(
    const float* __restrict__ pre1, const float* __restrict__ pre2,
    const uint32_t* __restrict__ w2f,
    const float* __restrict__ g1, const float* __restrict__ beta1,
    const float* __restrict__ b2,
    const float* __restrict__ g2, const float* __restrict__ beta2,
    const float* __restrict__ w3, const float* __restrict__ b3,
    float* __restrict__ logits) {

    __shared__ uint32_t aFrag[32][32][8];   // 32 K-chunks x 32 lanes x 8 words = 32 KB
    __shared__ float    h2buf[16][H2_];     // 32 KB

    const int lane = threadIdx.x & 31;
    const int wav  = threadIdx.x >> 5;      // 8 waves
    const int b    = blockIdx.x >> 4;
    const int s0   = (blockIdx.x & 15) << 4;

    // ---- Stage 1: h1 rows (wave w owns tile rows 2w, 2w+1) -------------------
    const float* p1 = pre1 + (size_t)b * H1_;
    for (int rr = 0; rr < 2; ++rr) {
        const int m = wav * 2 + rr;
        const float* p2 = pre2 + (size_t)(s0 + m) * H1_;
        float v0[16], v1[16], sum = 0.f, sq = 0.f;
        #pragma unroll
        for (int j = 0; j < 16; ++j) {
            int k = j * 64 + lane * 2;
            float a0 = p1[k] + p2[k];
            float a1 = p1[k + 1] + p2[k + 1];
            v0[j] = a0; v1[j] = a1;
            sum += a0 + a1; sq += a0 * a0 + a1 * a1;
        }
        #pragma unroll
        for (int off = 16; off; off >>= 1) {
            sum += __shfl_xor(sum, off, 32);
            sq  += __shfl_xor(sq,  off, 32);
        }
        float mean = sum * (1.f / 1024.f);
        float rstd = rsqrtf(sq * (1.f / 1024.f) - mean * mean + 1e-5f);
        // write LN+relu result as packed bf16 directly in A-fragment layout
        int ko   = (lane * 2) & 31;
        int vv   = (ko >> 1) & 7;
        int half = (ko >> 4) & 1;
        int fl   = m | (half << 4);
        #pragma unroll
        for (int j = 0; j < 16; ++j) {
            int k  = j * 64 + lane * 2;
            float h0 = fmaxf(fmaf((v0[j] - mean) * rstd, g1[k],     beta1[k]),     0.f);
            float h1 = fmaxf(fmaf((v1[j] - mean) * rstd, g1[k + 1], beta1[k + 1]), 0.f);
            int kc = j * 2 + (lane >> 4);
            aFrag[kc][fl][vv] = f2bf(h0) | (f2bf(h1) << 16);
        }
    }
    __syncthreads();

    // ---- Stage 2: 16x512 = A(16x1024,bf16) @ w2(1024x512,bf16) via WMMA ------
    union Frag { uint4 q[2]; v16bf v; };
    v8f acc[4] = {v8f{}, v8f{}, v8f{}, v8f{}};
    const int nt0 = wav * 4;                 // wave owns n-tiles nt0..nt0+3
    for (int kc = 0; kc < 32; ++kc) {
        Frag fa;
        fa.q[0] = *(const uint4*)&aFrag[kc][lane][0];
        fa.q[1] = *(const uint4*)&aFrag[kc][lane][4];
        #pragma unroll
        for (int t = 0; t < 4; ++t) {
            const uint32_t* bp = w2f + ((size_t)((kc * 32 + nt0 + t) * 32 + lane) << 3);
            Frag fb;
            fb.q[0] = *(const uint4*)(bp);
            fb.q[1] = *(const uint4*)(bp + 4);
            acc[t] = __builtin_amdgcn_wmma_f32_16x16x32_bf16(
                false, fa.v, false, fb.v, (short)0, acc[t], false, false);
        }
    }
    // scatter D tiles (+ b2) into LDS: D layout m = vv + (lane>>4)*8, n = lane&15
    {
        const int mb = (lane >> 4) << 3;
        #pragma unroll
        for (int t = 0; t < 4; ++t) {
            int n = (nt0 + t) * 16 + (lane & 15);
            float bias = b2[n];
            #pragma unroll
            for (int vv = 0; vv < 8; ++vv)
                h2buf[mb + vv][n] = acc[t][vv] + bias;
        }
    }
    __syncthreads();

    // ---- Stage 3: LN2 + relu + diagonal w3 dot -------------------------------
    for (int rr = 0; rr < 2; ++rr) {
        const int m = wav * 2 + rr;
        const int s = s0 + m;
        float vals[16], sum = 0.f, sq = 0.f;
        #pragma unroll
        for (int j = 0; j < 16; ++j) {
            float x = h2buf[m][j * 32 + lane];
            vals[j] = x; sum += x; sq += x * x;
        }
        #pragma unroll
        for (int off = 16; off; off >>= 1) {
            sum += __shfl_xor(sum, off, 32);
            sq  += __shfl_xor(sq,  off, 32);
        }
        float mean = sum * (1.f / 512.f);
        float rstd = rsqrtf(sq * (1.f / 512.f) - mean * mean + 1e-5f);
        float dot = 0.f;
        #pragma unroll
        for (int j = 0; j < 16; ++j) {
            int k = j * 32 + lane;
            float h = fmaxf(fmaf((vals[j] - mean) * rstd, g2[k], beta2[k]), 0.f);
            dot = fmaf(h, w3[(size_t)k * S_ + s], dot);
        }
        #pragma unroll
        for (int off = 16; off; off >>= 1) dot += __shfl_xor(dot, off, 32);
        if (lane == 0) logits[(size_t)b * S_ + s] = dot + b3[s];
    }
}

// ---------------------------------------------------------------------------
// Softmax over S + argsort(-logits) via rank counting (stable, ties by index)
// ---------------------------------------------------------------------------
__global__ void softmax_rank(const float* __restrict__ logits,
                             float* __restrict__ probs, float* __restrict__ ranking) {
    __shared__ float l[256];
    __shared__ float red[256];
    const int b = blockIdx.x, i = threadIdx.x;
    float li = logits[(size_t)b * S_ + i];
    l[i] = li;  red[i] = li;
    __syncthreads();
    for (int off = 128; off; off >>= 1) {
        if (i < off) red[i] = fmaxf(red[i], red[i + off]);
        __syncthreads();
    }
    float mx = red[0];
    __syncthreads();
    float e = __expf(li - mx);
    red[i] = e;
    __syncthreads();
    for (int off = 128; off; off >>= 1) {
        if (i < off) red[i] += red[i + off];
        __syncthreads();
    }
    probs[(size_t)b * S_ + i] = e / red[0];
    int rank = 0;
    for (int j = 0; j < 256; ++j) {
        float lj = l[j];
        rank += (lj > li) || (lj == li && j < i);
    }
    ranking[(size_t)b * S_ + rank] = (float)i;
}

// ---------------------------------------------------------------------------
extern "C" void kernel_launch(void* const* d_in, const int* in_sizes, int n_in,
                              void* d_out, int out_size, void* d_ws, size_t ws_size,
                              hipStream_t stream) {
    const float* fused = (const float*)d_in[0];
    const float* emb   = (const float*)d_in[1];
    const float* w1    = (const float*)d_in[2];
    const float* b1    = (const float*)d_in[3];
    const float* g1    = (const float*)d_in[4];
    const float* beta1 = (const float*)d_in[5];
    const float* w2    = (const float*)d_in[6];
    const float* b2    = (const float*)d_in[7];
    const float* g2    = (const float*)d_in[8];
    const float* beta2 = (const float*)d_in[9];
    const float* w3    = (const float*)d_in[10];
    const float* b3    = (const float*)d_in[11];
    float* out = (float*)d_out;   // [logits | probs | ranking], each B_*S_

    float*    pre1 = (float*)d_ws;                       // 1 MB
    float*    pre2 = pre1 + (size_t)B_ * H1_;            // 1 MB
    uint32_t* w2f  = (uint32_t*)(pre2 + (size_t)S_ * H1_); // 1 MB (bf16 pairs)

    prep_pre1 <<<(B_ * H1_) / 256, 256, 0, stream>>>(fused, w1, b1, pre1);
    prep_pre2 <<<(S_ * H1_) / 256, 256, 0, stream>>>(emb, w1, pre2);
    prep_w2frag<<<(H1_ * H2_ / 2) / 256, 256, 0, stream>>>(w2, w2f);

    rca_main<<<B_ * (S_ / 16), 256, 0, stream>>>(pre1, pre2, w2f,
                                                 g1, beta1, b2, g2, beta2, w3, b3,
                                                 out);
    softmax_rank<<<B_, 256, 0, stream>>>(out, out + (size_t)B_ * S_,
                                         out + 2 * (size_t)B_ * S_);
}
